// AttentativeResidual_86225763435204
// MI455X (gfx1250) — matrix operand order repaired and checked by reading, not compiled, not run on hardware
//
#include <hip/hip_runtime.h>

// MI455X / gfx1250, wave32. bf16 WMMA (f32 accum) flash-attention.
//
// ws layout: Kbf [B][M][128] bf16 (2 MB) | Vt [B][128][M] bf16 (2 MB)

typedef __attribute__((ext_vector_type(16))) __bf16 v16bf;
typedef __attribute__((ext_vector_type(8)))  __bf16 v8bf;
typedef __attribute__((ext_vector_type(8)))  float  v8f;

constexpr int DH  = 128;   // head dim / channel
constexpr int MK  = 2048;  // keys per batch
constexpr int NQ  = 2048;  // queries per (b,t)
constexpr int TT  = 8;     // T
constexpr int BB  = 4;     // B
constexpr int WTS = 144;   // padded LDS row stride for transposed W (bf16 elems)
constexpr int KT  = 64;    // keys per flash-loop iteration

__device__ __forceinline__ __bf16 to_bf16(float f) { return (__bf16)f; }

// Stage W(128x128 f32, global) into LDS transposed as bf16: Wt[d][c], row
// stride WTS. Coalesced b64 global reads; 2-byte LDS stores (one-time cost).
__device__ __forceinline__ void stage_wt(const float* __restrict__ W,
                                         __bf16* __restrict__ Wt, int tid)
{
#pragma unroll
    for (int i = 0; i < 32; ++i) {
        const int p   = i * 256 + tid;     // float2 index
        const int idx = p * 2;
        const int c = idx >> 7;
        const int d = idx & 127;
        const float2 w = *(const float2*)(W + idx);
        Wt[(size_t)d * WTS + c]       = to_bf16(w.x);
        Wt[(size_t)(d + 1) * WTS + c] = to_bf16(w.y);
    }
}

// A-frag (16-bit 16x32): lane row = lane&15; elem e -> k = 16*hf + e
__device__ __forceinline__ void load_a_frags(const float* __restrict__ rows,
                                             v16bf a[4], int lane)
{
    const int rlo = lane & 15;
    const int hf  = lane >> 4;
#pragma unroll
    for (int jc = 0; jc < 4; ++jc) {
        const float* base = rows + (size_t)rlo * DH + 32 * jc;
#pragma unroll
        for (int e = 0; e < 8; ++e) {
            a[jc][e]     = to_bf16(base[16 * hf + e]);
            a[jc][e + 8] = to_bf16(base[16 * hf + 8 + e]);
        }
    }
}

// acc[jd] = 16x16 C-frag of rows(16x128) @ W(128x128), W staged transposed in
// LDS. B-frag (16-bit 32x16): lane col = lane&15; elem e -> k = 16*hf + e,
// i.e. Wt[16*jd + col][32*jc + 16*hf + e] -> one contiguous 32B LDS read.
__device__ __forceinline__ void matmul_a_wt(const v16bf a[4],
                                            const __bf16* __restrict__ Wt,
                                            v8f acc[8], int lane)
{
    const int n16 = lane & 15;
    const int hf  = lane >> 4;
#pragma unroll
    for (int jd = 0; jd < 8; ++jd) {
        v8f c = {};
#pragma unroll
        for (int jc = 0; jc < 4; ++jc) {
            const v16bf bfrag =
                *(const v16bf*)(Wt + (size_t)(16 * jd + n16) * WTS + 32 * jc + 16 * hf);
            c = __builtin_amdgcn_wmma_f32_16x16x32_bf16(false, a[jc], false, bfrag,
                                                        (short)0, c, false, false);
        }
        acc[jd] = c;
    }
}

// K = rs@Wk -> Kbf[B][M][DH];  V = rs@Wv -> Vt[B][DH][M] (transposed)
__global__ __launch_bounds__(256) void proj_kv_kernel(
    const float* __restrict__ rs, const float* __restrict__ Wk,
    const float* __restrict__ Wv, __bf16* __restrict__ Kbf,
    __bf16* __restrict__ Vt)
{
    __shared__ __bf16 wt[128 * WTS];   // 36 KB

    const int wave = threadIdx.x >> 5;
    const int lane = threadIdx.x & 31;
    const int n16  = lane & 15;
    const int hf   = lane >> 4;
    const int mpb  = MK / 128;              // 16
    const int b    = blockIdx.x / mpb;
    const int mblk = blockIdx.x % mpb;
    const int mbase = mblk * 128 + wave * 16;
    const float* rows = rs + ((size_t)b * MK + mbase) * DH;

    v16bf a[4];
    load_a_frags(rows, a, lane);

    v8f acc[8];
    // ---- K projection ----
    stage_wt(Wk, wt, threadIdx.x);
    __syncthreads();
    matmul_a_wt(a, wt, acc, lane);
#pragma unroll
    for (int jd = 0; jd < 8; ++jd)
#pragma unroll
        for (int r = 0; r < 8; ++r) {
            const int row = mbase + r + 8 * hf;
            Kbf[((size_t)b * MK + row) * DH + 16 * jd + n16] = to_bf16(acc[jd][r]);
        }
    __syncthreads();
    // ---- V projection (stored transposed) ----
    stage_wt(Wv, wt, threadIdx.x);
    __syncthreads();
    matmul_a_wt(a, wt, acc, lane);
#pragma unroll
    for (int jd = 0; jd < 8; ++jd)
#pragma unroll
        for (int r = 0; r < 8; ++r) {
            const int row = mbase + r + 8 * hf;
            Vt[((size_t)b * DH + 16 * jd + n16) * MK + row] = to_bf16(acc[jd][r]);
        }
}

// Fused Q-projection + flash attention + residual add.
__global__ __launch_bounds__(256) void attn_kernel(
    const float* __restrict__ x,  const float* __restrict__ Wq,
    const __bf16* __restrict__ Kbf, const __bf16* __restrict__ Vt,
    float* __restrict__ out)
{
    __shared__ union {                    // 36 KB, phases don't overlap
        __bf16 wt[128 * WTS];             //   phase 1: transposed Wq
        __bf16 q[8][16][DH];              //   phase 2: Q C-frag -> A-frag
    } st;
    __shared__ __bf16 ptile[8][16][KT];   // 16 KB: P relayout per key block

    const int wave = threadIdx.x >> 5;
    const int lane = threadIdx.x & 31;
    const int n16  = lane & 15;
    const int hf   = lane >> 4;
    const int bt   = blockIdx.y;          // b*T + t
    const int b    = bt >> 3;             // T = 8
    const int qbase = blockIdx.x * 128 + wave * 16;

    const float* xrows = x + ((size_t)bt * NQ + qbase) * DH;

    // ---- Q = X @ Wq (WMMA) via LDS-staged Wq^T ----
    v16bf xa[4];
    load_a_frags(xrows, xa, lane);
    stage_wt(Wq, st.wt, threadIdx.x);
    __syncthreads();
    v8f qacc[8];
    matmul_a_wt(xa, st.wt, qacc, lane);
    __syncthreads();                      // done reading wt, reuse as q stage

    // stage Q into LDS (bf16) and reload in A layout
#pragma unroll
    for (int jd = 0; jd < 8; ++jd)
#pragma unroll
        for (int r = 0; r < 8; ++r)
            st.q[wave][r + 8 * hf][16 * jd + n16] = to_bf16(qacc[jd][r]);
    __syncthreads();

    v16bf qa[4];
    {
        const __bf16* qrow = &st.q[wave][n16][0];
#pragma unroll
        for (int jc = 0; jc < 4; ++jc) {
            const v8bf lo = *(const v8bf*)(qrow + 32 * jc + 16 * hf);
            const v8bf hi = *(const v8bf*)(qrow + 32 * jc + 16 * hf + 8);
#pragma unroll
            for (int e = 0; e < 8; ++e) { qa[jc][e] = lo[e]; qa[jc][e + 8] = hi[e]; }
        }
    }

    // ---- online-softmax state (rows r+8*hf live in this lane's half) ----
    v8f o[8];
#pragma unroll
    for (int jd = 0; jd < 8; ++jd)
#pragma unroll
        for (int e = 0; e < 8; ++e) o[jd][e] = 0.0f;
    float mrow[8], lrow[8];
#pragma unroll
    for (int r = 0; r < 8; ++r) { mrow[r] = -1e30f; lrow[r] = 0.0f; }

    const __bf16* Kb = Kbf + (size_t)b * MK * DH;
    const __bf16* Vb = Vt  + (size_t)b * DH * MK;

    for (int kb = 0; kb < MK / KT; ++kb) {
        // S = Q K^T for 64 keys -> four 16x16 f32 C-frags
        v8f s[4];
#pragma unroll
        for (int nb = 0; nb < 4; ++nb) {
            const int key = kb * KT + 16 * nb + n16;
            const __bf16* krow = Kb + (size_t)key * DH;
            v8f c = {};
#pragma unroll
            for (int jc = 0; jc < 4; ++jc) {
                const v16bf bfrag = *(const v16bf*)(krow + 32 * jc + 16 * hf);
                c = __builtin_amdgcn_wmma_f32_16x16x32_bf16(false, qa[jc], false, bfrag,
                                                            (short)0, c, false, false);
            }
            s[nb] = c;
        }
        if (kb + 1 < MK / KT) {   // global_prefetch_b8 next K / V tiles
            __builtin_prefetch(Kb + (size_t)(kb * KT + KT + lane * 2) * DH, 0, 3);
            __builtin_prefetch(Vb + (size_t)(lane * 4) * MK + kb * KT + KT, 0, 3);
        }

        // online softmax per row (width-16 reductions match C-frag striping)
        float alpha[8];
#pragma unroll
        for (int r = 0; r < 8; ++r) {
            float v = fmaxf(fmaxf(s[0][r], s[1][r]), fmaxf(s[2][r], s[3][r]));
#pragma unroll
            for (int off = 8; off >= 1; off >>= 1)
                v = fmaxf(v, __shfl_xor(v, off, 16));
            const float mnew = fmaxf(mrow[r], v);
            alpha[r] = __expf(mrow[r] - mnew);
            mrow[r]  = mnew;
            float ps = 0.0f;
#pragma unroll
            for (int nb = 0; nb < 4; ++nb) {
                const float p = __expf(s[nb][r] - mnew);
                s[nb][r] = p;
                ps += p;
            }
#pragma unroll
            for (int off = 8; off >= 1; off >>= 1)
                ps += __shfl_xor(ps, off, 16);
            lrow[r] = lrow[r] * alpha[r] + ps;
        }
#pragma unroll
        for (int jd = 0; jd < 8; ++jd)
#pragma unroll
            for (int r = 0; r < 8; ++r) o[jd][r] *= alpha[r];

        // stage P (16x64) to LDS, reload as two bf16 A-frags
#pragma unroll
        for (int nb = 0; nb < 4; ++nb)
#pragma unroll
            for (int r = 0; r < 8; ++r)
                ptile[wave][r + 8 * hf][16 * nb + n16] = to_bf16(s[nb][r]);
        v16bf pa[2];
        {
            const __bf16* prow = &ptile[wave][n16][0];
#pragma unroll
            for (int jh = 0; jh < 2; ++jh) {
                const v8bf lo = *(const v8bf*)(prow + 32 * jh + 16 * hf);
                const v8bf hi = *(const v8bf*)(prow + 32 * jh + 16 * hf + 8);
#pragma unroll
                for (int e = 0; e < 8; ++e) { pa[jh][e] = lo[e]; pa[jh][e + 8] = hi[e]; }
            }
        }

        // O += P @ V  (Vt rows are contiguous over keys -> clean B-frag loads)
#pragma unroll
        for (int jd = 0; jd < 8; ++jd) {
            const __bf16* vrow = Vb + (size_t)(16 * jd + n16) * MK + kb * KT;
            const v16bf vf0 = *(const v16bf*)(vrow + 16 * hf);
            o[jd] = __builtin_amdgcn_wmma_f32_16x16x32_bf16(false, pa[0], false, vf0,
                                                            (short)0, o[jd], false, false);
            const v16bf vf1 = *(const v16bf*)(vrow + 32 + 16 * hf);
            o[jd] = __builtin_amdgcn_wmma_f32_16x16x32_bf16(false, pa[1], false, vf1,
                                                            (short)0, o[jd], false, false);
        }
    }

    // ---- epilogue: out = x + O / l ----
#pragma unroll
    for (int jd = 0; jd < 8; ++jd)
#pragma unroll
        for (int r = 0; r < 8; ++r) {
            const size_t idx =
                ((size_t)bt * NQ + qbase + r + 8 * hf) * DH + 16 * jd + n16;
            out[idx] = x[idx] + o[jd][r] / lrow[r];
        }
}

extern "C" void kernel_launch(void* const* d_in, const int* in_sizes, int n_in,
                              void* d_out, int out_size, void* d_ws, size_t ws_size,
                              hipStream_t stream) {
    const float* x  = (const float*)d_in[0];   // [B,T,N,C]
    const float* rs = (const float*)d_in[1];   // [B,M,C]
    const float* Wq = (const float*)d_in[2];
    const float* Wk = (const float*)d_in[3];
    const float* Wv = (const float*)d_in[4];
    float* out = (float*)d_out;

    __bf16* Kbf = (__bf16*)d_ws;                         // [B][M][DH] bf16
    __bf16* Vtr = Kbf + (size_t)BB * MK * DH;            // [B][DH][M] bf16

    proj_kv_kernel<<<dim3(BB * (MK / 128)), dim3(256), 0, stream>>>(rs, Wk, Wv, Kbf, Vtr);
    attn_kernel<<<dim3(NQ / 128, BB * TT), dim3(256), 0, stream>>>(x, Wq, Kbf, Vtr, out);
}
